// UNet_AttentionBlock_72292889526976
// MI455X (gfx1250) — matrix operand order repaired and checked by reading, not compiled, not run on hardware
//
#include <hip/hip_runtime.h>
#include <hip/hip_bf16.h>
#include <stdint.h>

typedef __attribute__((ext_vector_type(16))) _Float16 v16h;
typedef __attribute__((ext_vector_type(8)))  _Float16 v8h;
typedef __attribute__((ext_vector_type(8)))  float    v8f;

#define BATCH   16
#define CCH     256
#define LSEQ    1024      // 32*32 spatial
#define HEADS   8
#define HDIM    32
#define NQKV    768
#define DCOND   128
#define GROUPS  32
#define GN_EPS  1e-5f

// ---------------------------------------------------------------------------
// K0: f32 -> f16 conversion (weights)
// ---------------------------------------------------------------------------
__global__ void cvt_half_kernel(const float* __restrict__ src,
                                _Float16* __restrict__ dst, int n) {
    int i = blockIdx.x * blockDim.x + threadIdx.x;
    if (i < n) dst[i] = (_Float16)src[i];
}

// ---------------------------------------------------------------------------
// K1: conditioning embeddings: emb[b][c] = sum of 3 (silu(x) @ W.T + b)
// grid(B), block(256): thread = output channel
// ---------------------------------------------------------------------------
__global__ void cond_embed_kernel(const float* __restrict__ t,
                                  const float* __restrict__ cond,
                                  const float* __restrict__ cond2,
                                  const float* __restrict__ tw, const float* __restrict__ tb,
                                  const float* __restrict__ cw, const float* __restrict__ cb,
                                  const float* __restrict__ c2w, const float* __restrict__ c2b,
                                  float* __restrict__ emb) {
    int b = blockIdx.x, c = threadIdx.x;
    float acc = tb[c] + cb[c] + c2b[c];
    const float* tp  = t     + b * DCOND;
    const float* cp  = cond  + b * DCOND;
    const float* c2p = cond2 + b * DCOND;
    const float* twp  = tw  + c * DCOND;
    const float* cwp  = cw  + c * DCOND;
    const float* c2wp = c2w + c * DCOND;
    for (int k = 0; k < DCOND; ++k) {
        float a = tp[k];  acc += (a / (1.f + __expf(-a))) * twp[k];
        float d = cp[k];  acc += (d / (1.f + __expf(-d))) * cwp[k];
        float e = c2p[k]; acc += (e / (1.f + __expf(-e))) * c2wp[k];
    }
    emb[b * CCH + c] = acc;
}

// ---------------------------------------------------------------------------
// K2: groupnorm stats. grid(B*GROUPS), block(256). 8 ch * 1024 = 8192 elems/group
// ---------------------------------------------------------------------------
__global__ void gn_stats_kernel(const float* __restrict__ x,
                                float* __restrict__ mean, float* __restrict__ rstd) {
    int bg = blockIdx.x;
    int b = bg >> 5, g = bg & 31;
    const float* p = x + ((size_t)b * CCH + g * 8) * LSEQ;
    float s = 0.f, s2 = 0.f;
    for (int i = threadIdx.x; i < 8 * LSEQ; i += 256) {
        float v = p[i]; s += v; s2 += v * v;
    }
    __shared__ float l1[256], l2[256];
    int tid = threadIdx.x;
    l1[tid] = s; l2[tid] = s2;
    __syncthreads();
    for (int off = 128; off > 0; off >>= 1) {
        if (tid < off) { l1[tid] += l1[tid + off]; l2[tid] += l2[tid + off]; }
        __syncthreads();
    }
    if (tid == 0) {
        float mn  = l1[0] * (1.f / 8192.f);
        float var = l2[0] * (1.f / 8192.f) - mn * mn;
        mean[bg] = mn;
        rstd[bg] = rsqrtf(var + GN_EPS);
    }
}

// ---------------------------------------------------------------------------
// K3: normalize + add emb, transpose (b,c,l) -> (b,l,c), f32 + f16 copies.
// grid(B*CCH), block(256)
// ---------------------------------------------------------------------------
__global__ void gn_apply_kernel(const float* __restrict__ x,
                                const float* __restrict__ mean, const float* __restrict__ rstd,
                                const float* __restrict__ gw, const float* __restrict__ gb,
                                const float* __restrict__ emb,
                                float* __restrict__ xs_f, _Float16* __restrict__ xs_h) {
    int bc = blockIdx.x;
    int b = bc >> 8, c = bc & 255;
    int g = c >> 3;
    float m = mean[b * GROUPS + g], r = rstd[b * GROUPS + g];
    float scale = r * gw[c];
    float shift = gb[c] + emb[b * CCH + c] - m * scale;
    const float* px = x + (size_t)bc * LSEQ;
    for (int l = threadIdx.x; l < LSEQ; l += 256) {
        float v = px[l] * scale + shift;
        size_t o = ((size_t)b * LSEQ + l) * CCH + c;
        xs_f[o] = v;
        xs_h[o] = (_Float16)v;
    }
}

// ---------------------------------------------------------------------------
// WMMA fragment helpers (this code's contiguous k order for A / B; ISA 32-bit C layout, wave32)
//   A (16xK=32): lane row = lane&15; elem i: K = 16*(lane>>4) + i
//   B (K=32x16): lane col = lane&15; elem i: K = 16*(lane>>4) + i
//   C (16x16):   lane col = lane&15; elem r: row = r + 8*(lane>>4)
// ---------------------------------------------------------------------------
__device__ inline v16h load_a_frag(const _Float16* base, int stride_row,
                                   int ln, int khi, int k0) {
    const _Float16* p = base + (size_t)ln * stride_row + k0 + 16 * khi;
    v8h lo = *(const v8h*)(p);
    v8h hi = *(const v8h*)(p + 8);
    return __builtin_shufflevector(lo, hi, 0, 1, 2, 3, 4, 5, 6, 7,
                                           8, 9, 10, 11, 12, 13, 14, 15);
}

// ---------------------------------------------------------------------------
// K4: QKV GEMM: (b,L,256) x Wqkv(768,256)^T -> q/k/vT f16 buffers per head.
// grid(12, 8, 16) = (ntile64, mtile128, b), block(256)=8 waves, wave: 16x64 tile
// ---------------------------------------------------------------------------
__global__ void qkv_gemm_kernel(const _Float16* __restrict__ xs_h,
                                const _Float16* __restrict__ wh,   // (768,256) row-major
                                const float* __restrict__ bias,
                                _Float16* __restrict__ qh,   // (b,h,L,32), pre-scaled
                                _Float16* __restrict__ kh,   // (b,h,L,32)
                                _Float16* __restrict__ vth)  // (b,h,32,L)
{
    int b = blockIdx.z;
    int n0 = blockIdx.x * 64, m0 = blockIdx.y * 128;
    int wave = threadIdx.x >> 5, lane = threadIdx.x & 31;
    int khi = lane >> 4, ln = lane & 15;
    int row = m0 + wave * 16;
    const _Float16* A = xs_h + ((size_t)b * LSEQ + row) * CCH;

    v8f acc[4] = {};
    for (int k0 = 0; k0 < CCH; k0 += 32) {
        v16h a = load_a_frag(A, CCH, ln, khi, k0);
        #pragma unroll
        for (int t = 0; t < 4; ++t) {
            int col = n0 + t * 16 + ln;
            v16h bf = *(const v16h*)(wh + (size_t)col * CCH + k0 + 16 * khi);
            acc[t] = __builtin_amdgcn_wmma_f32_16x16x32_f16(
                false, a, false, bf, (short)0, acc[t], false, false);
        }
    }
    #pragma unroll
    for (int t = 0; t < 4; ++t) {
        int col = n0 + t * 16 + ln;
        int head = col / 96, rem = col - head * 96;
        float bv = bias[col];
        size_t bh = (size_t)b * HEADS + head;
        #pragma unroll
        for (int r = 0; r < 8; ++r) {
            int l = row + r + 8 * khi;
            float v = acc[t][r] + bv;
            if (rem < 32)
                qh[(bh * LSEQ + l) * HDIM + rem] = (_Float16)(v * 0.0625f); // *1/sqrt(256)
            else if (rem < 64)
                kh[(bh * LSEQ + l) * HDIM + (rem - 32)] = (_Float16)v;
            else
                vth[(bh * HDIM + (rem - 64)) * LSEQ + l] = (_Float16)v;
        }
    }
}

// ---------------------------------------------------------------------------
// K5: flash attention. grid(8, HEADS, B) = (qchunk128, h, b), block(256)=8 waves.
// Wave owns 16 queries; loops keys in 32-chunks. Computes S^T = K·Q^T so the
// WMMA C-layout of S^T is exactly the A-layout needed for P·V (no shuffle).
// ---------------------------------------------------------------------------
__global__ void flash_attn_kernel(const _Float16* __restrict__ qh,
                                  const _Float16* __restrict__ kh,
                                  const _Float16* __restrict__ vth,
                                  _Float16* __restrict__ attn_h) { // (b,L,256)
    int b = blockIdx.z, h = blockIdx.y;
    size_t bh = (size_t)b * HEADS + h;
    int wave = threadIdx.x >> 5, lane = threadIdx.x & 31;
    int khi = lane >> 4, ln = lane & 15;
    int q0 = blockIdx.x * 128 + wave * 16;

    const _Float16* Q  = qh  + bh * LSEQ * HDIM;
    const _Float16* K  = kh  + bh * LSEQ * HDIM;
    const _Float16* VT = vth + bh * HDIM * LSEQ;

    // Q^T as B fragment (fixed over the key loop): col = query ln, K(i)=16*khi+i
    v16h qB = *(const v16h*)(Q + (size_t)(q0 + ln) * HDIM + 16 * khi);

    float m = -3.0e38f, lsum = 0.f;
    v8f acc0 = {}, acc1 = {};
    const v8f zc = {};

    for (int kc = 0; kc < LSEQ; kc += 32) {
        // K chunk as two A fragments (16 keys x 32 dims each)
        v16h kA0 = load_a_frag(K + (size_t)kc * HDIM,        HDIM, ln, khi, 0);
        v16h kA1 = load_a_frag(K + (size_t)(kc + 16) * HDIM, HDIM, ln, khi, 0);

        v8f st0 = __builtin_amdgcn_wmma_f32_16x16x32_f16(
            false, kA0, false, qB, (short)0, zc, false, false);
        v8f st1 = __builtin_amdgcn_wmma_f32_16x16x32_f16(
            false, kA1, false, qB, (short)0, zc, false, false);
        // st0[r] = S[query q0+ln][key kc + r + 8*khi]; st1: keys +16

        // ---- online softmax (per-lane = per-query column) ----
        float cmax = st0[0];
        #pragma unroll
        for (int r = 1; r < 8; ++r) cmax = fmaxf(cmax, st0[r]);
        #pragma unroll
        for (int r = 0; r < 8; ++r) cmax = fmaxf(cmax, st1[r]);
        cmax = fmaxf(cmax, __shfl_xor(cmax, 16, 32));
        float mnew = fmaxf(m, cmax);
        float alpha = __expf(m - mnew);

        float p0[8], p1[8], csum = 0.f;
        #pragma unroll
        for (int r = 0; r < 8; ++r) {
            p0[r] = __expf(st0[r] - mnew); csum += p0[r];
            p1[r] = __expf(st1[r] - mnew); csum += p1[r];
        }
        csum += __shfl_xor(csum, 16, 32);
        lsum = lsum * alpha + csum;
        m = mnew;

        // P as A fragment: elem i<8 -> key i+8khi (=st0[i]); i>=8 -> key 16+(i-8)+8khi (=st1[i-8])
        v16h pA;
        #pragma unroll
        for (int i = 0; i < 8; ++i) {
            pA[i]     = (_Float16)p0[i];
            pA[i + 8] = (_Float16)p1[i];
        }

        // V chunk as two B fragments: vB0 dims 0..15, vB1 dims 16..31
        const _Float16* vp0 = VT + (size_t)ln * LSEQ        + kc + 8 * khi;
        const _Float16* vp1 = VT + (size_t)(16 + ln) * LSEQ + kc + 8 * khi;
        v16h vB0 = __builtin_shufflevector(*(const v8h*)vp0, *(const v8h*)(vp0 + 16), 0, 1, 2, 3, 4, 5, 6, 7, 8, 9, 10, 11, 12, 13, 14, 15);
        v16h vB1 = __builtin_shufflevector(*(const v8h*)vp1, *(const v8h*)(vp1 + 16), 0, 1, 2, 3, 4, 5, 6, 7, 8, 9, 10, 11, 12, 13, 14, 15);

        // rescale accumulators: acc row r -> query r+8*khi; fetch that query's alpha
        #pragma unroll
        for (int r = 0; r < 8; ++r) {
            float ar = __shfl(alpha, r + 8 * khi, 32);
            acc0[r] *= ar;
            acc1[r] *= ar;
        }
        acc0 = __builtin_amdgcn_wmma_f32_16x16x32_f16(
            false, pA, false, vB0, (short)0, acc0, false, false);
        acc1 = __builtin_amdgcn_wmma_f32_16x16x32_f16(
            false, pA, false, vB1, (short)0, acc1, false, false);
    }

    // finalize: divide each row by its softmax sum, write (b, l, h*32 + d) f16
    #pragma unroll
    for (int r = 0; r < 8; ++r) {
        float lr = __shfl(lsum, r + 8 * khi, 32);
        float inv = 1.0f / lr;
        int l = q0 + r + 8 * khi;
        size_t o = ((size_t)b * LSEQ + l) * CCH + h * HDIM;
        attn_h[o + ln]      = (_Float16)(acc0[r] * inv);
        attn_h[o + 16 + ln] = (_Float16)(acc1[r] * inv);
    }
}

// ---------------------------------------------------------------------------
// K6: output projection + bias + residual, transpose back to (b,c,h,w).
// grid(4, 8, 16) = (ntile64, mtile128, b), block(256)=8 waves
// ---------------------------------------------------------------------------
__global__ void out_gemm_kernel(const _Float16* __restrict__ attn_h,
                                const _Float16* __restrict__ wh,  // (256,256) row-major
                                const float* __restrict__ bias,
                                const float* __restrict__ xs_f,
                                float* __restrict__ out) {
    int b = blockIdx.z;
    int n0 = blockIdx.x * 64, m0 = blockIdx.y * 128;
    int wave = threadIdx.x >> 5, lane = threadIdx.x & 31;
    int khi = lane >> 4, ln = lane & 15;
    int row = m0 + wave * 16;
    const _Float16* A = attn_h + ((size_t)b * LSEQ + row) * CCH;

    v8f acc[4] = {};
    for (int k0 = 0; k0 < CCH; k0 += 32) {
        v16h a = load_a_frag(A, CCH, ln, khi, k0);
        #pragma unroll
        for (int t = 0; t < 4; ++t) {
            int col = n0 + t * 16 + ln;
            v16h bf = *(const v16h*)(wh + (size_t)col * CCH + k0 + 16 * khi);
            acc[t] = __builtin_amdgcn_wmma_f32_16x16x32_f16(
                false, a, false, bf, (short)0, acc[t], false, false);
        }
    }
    #pragma unroll
    for (int t = 0; t < 4; ++t) {
        int c = n0 + t * 16 + ln;
        float bv = bias[c];
        #pragma unroll
        for (int r = 0; r < 8; ++r) {
            int l = row + r + 8 * khi;
            float v = acc[t][r] + bv + xs_f[((size_t)b * LSEQ + l) * CCH + c];
            out[((size_t)b * CCH + c) * LSEQ + l] = v;
        }
    }
}

// ---------------------------------------------------------------------------
// Host launch
// ---------------------------------------------------------------------------
static inline size_t align256(size_t x) { return (x + 255) & ~(size_t)255; }

extern "C" void kernel_launch(void* const* d_in, const int* in_sizes, int n_in,
                              void* d_out, int out_size, void* d_ws, size_t ws_size,
                              hipStream_t stream) {
    const float* x      = (const float*)d_in[0];
    const float* t      = (const float*)d_in[1];
    const float* cond   = (const float*)d_in[2];
    const float* cond2  = (const float*)d_in[3];
    const float* gn_w   = (const float*)d_in[4];
    const float* gn_b   = (const float*)d_in[5];
    const float* proj_w = (const float*)d_in[6];
    const float* proj_b = (const float*)d_in[7];
    const float* out_w  = (const float*)d_in[8];
    const float* out_b  = (const float*)d_in[9];
    const float* time_w = (const float*)d_in[10];
    const float* time_b = (const float*)d_in[11];
    const float* cond_w = (const float*)d_in[12];
    const float* cond_b = (const float*)d_in[13];
    const float* cond2_w = (const float*)d_in[14];
    const float* cond2_b = (const float*)d_in[15];
    float* out = (float*)d_out;

    // workspace carve-out
    char* ws = (char*)d_ws;
    size_t off = 0;
    float* emb = (float*)(ws + off);            off = align256(off + BATCH * CCH * 4);
    float* mean = (float*)(ws + off);           off = align256(off + BATCH * GROUPS * 4);
    float* rstd = (float*)(ws + off);           off = align256(off + BATCH * GROUPS * 4);
    float* xs_f = (float*)(ws + off);           off = align256(off + (size_t)BATCH * LSEQ * CCH * 4);
    _Float16* xs_h = (_Float16*)(ws + off);     off = align256(off + (size_t)BATCH * LSEQ * CCH * 2);
    _Float16* proj_wh = (_Float16*)(ws + off);  off = align256(off + (size_t)NQKV * CCH * 2);
    _Float16* out_wh = (_Float16*)(ws + off);   off = align256(off + (size_t)CCH * CCH * 2);
    _Float16* qh = (_Float16*)(ws + off);       off = align256(off + (size_t)BATCH * HEADS * LSEQ * HDIM * 2);
    _Float16* kh = (_Float16*)(ws + off);       off = align256(off + (size_t)BATCH * HEADS * LSEQ * HDIM * 2);
    _Float16* vth = (_Float16*)(ws + off);      off = align256(off + (size_t)BATCH * HEADS * HDIM * LSEQ * 2);
    _Float16* attn_h = (_Float16*)(ws + off);   off = align256(off + (size_t)BATCH * LSEQ * CCH * 2);
    (void)ws_size; (void)n_in; (void)in_sizes; (void)out_size;

    // weight conversion
    {
        int n1 = NQKV * CCH;
        cvt_half_kernel<<<(n1 + 255) / 256, 256, 0, stream>>>(proj_w, proj_wh, n1);
        int n2 = CCH * CCH;
        cvt_half_kernel<<<(n2 + 255) / 256, 256, 0, stream>>>(out_w, out_wh, n2);
    }
    // conditioning embeddings
    cond_embed_kernel<<<BATCH, CCH, 0, stream>>>(t, cond, cond2,
                                                 time_w, time_b, cond_w, cond_b,
                                                 cond2_w, cond2_b, emb);
    // groupnorm
    gn_stats_kernel<<<BATCH * GROUPS, 256, 0, stream>>>(x, mean, rstd);
    gn_apply_kernel<<<BATCH * CCH, 256, 0, stream>>>(x, mean, rstd, gn_w, gn_b,
                                                     emb, xs_f, xs_h);
    // QKV projection (WMMA)
    qkv_gemm_kernel<<<dim3(NQKV / 64, LSEQ / 128, BATCH), 256, 0, stream>>>(
        xs_h, proj_wh, proj_b, qh, kh, vth);
    // flash attention (WMMA)
    flash_attn_kernel<<<dim3(LSEQ / 128, HEADS, BATCH), 256, 0, stream>>>(
        qh, kh, vth, attn_h);
    // output projection + residual + transpose (WMMA)
    out_gemm_kernel<<<dim3(CCH / 64, LSEQ / 128, BATCH), 256, 0, stream>>>(
        attn_h, out_wh, out_b, xs_f, out);
}
